// lstm_target_65085934404072
// MI455X (gfx1250) — compile-verified, matrix operand order checked by reading, not run on hardware
//
#include <hip/hip_runtime.h>
#include <hip/hip_bf16.h>

typedef __bf16 bf16_t;
typedef __attribute__((ext_vector_type(16))) __bf16 v16bf;
typedef __attribute__((ext_vector_type(8)))  float  v8f;
typedef __attribute__((ext_vector_type(4)))  int    v4i;

// ---------------------------------------------------------------------------
// Problem constants (from the reference)
// ---------------------------------------------------------------------------
#define BB   64
#define TT   256
#define HH   1024
#define DIN  3072        // H + 2E
#define FOURH 4096
#define MROWS (BB * TT)  // 16384

// ---------------------------------------------------------------------------
// CDNA5 async global->LDS staging (ASYNCcnt-tracked copy). Signature per the
// compiler diagnostic: (global v4i*, local v4i*, i32 offset, i32 cpol).
// Guarded so the kernel still compiles on toolchains without the builtin.
// ---------------------------------------------------------------------------
#if defined(__gfx1250__) && __has_builtin(__builtin_amdgcn_global_load_async_to_lds_b128)
#define USE_ASYNC_LDS 1
typedef __attribute__((address_space(1))) v4i g1v4i;
typedef __attribute__((address_space(3))) v4i l3v4i;
static __device__ __forceinline__ void async_copy16(const void* g, void* l) {
  void* gv = const_cast<void*>(g);
  __builtin_amdgcn_global_load_async_to_lds_b128((g1v4i*)gv, (l3v4i*)l, 0, 0);
}
static __device__ __forceinline__ void async_wait0() {
#if __has_builtin(__builtin_amdgcn_s_wait_asynccnt)
  __builtin_amdgcn_s_wait_asynccnt(0);
#else
  asm volatile("s_wait_asynccnt 0x0" ::: "memory");
#endif
}
#else
#define USE_ASYNC_LDS 0
#endif

// ---------------------------------------------------------------------------
// WMMA fragment gather from an LDS tile stored as tile[rc][32] (k contiguous).
// 16-bit A layout (ISA 7.12.2): lane<16 -> K {0..7,16..23}, lane>=16 -> +8.
// B uses the mirrored layout with rc = column.
// ---------------------------------------------------------------------------
static __device__ __forceinline__ v16bf frag_from_lds(const bf16_t* base, int rc, int lane) {
  v16bf f;
  const int khalf = (lane & 16) ? 8 : 0;
#pragma unroll
  for (int v = 0; v < 8; ++v) {
    const int k = ((v < 4) ? (2 * v) : (16 + 2 * (v - 4))) + khalf;
    f[2 * v]     = base[rc * 32 + k];
    f[2 * v + 1] = base[rc * 32 + k + 1];
  }
  return f;
}

static __device__ __forceinline__ v8f wmma_bf16(const v16bf a, const v16bf b, v8f c) {
  return __builtin_amdgcn_wmma_f32_16x16x32_bf16(false, a, false, b, (short)0, c, false, false);
}

// ---------------------------------------------------------------------------
// Tiled WMMA GEMM: C[M,N] = A[M,K](bf16,row-major) @ W[K,N](bf16,row-major) + bias[N]
// Block tile 64x128, 256 threads = 8 waves in a 2x4 grid; each wave owns a
// 32x32 micro-tile: 2 A frags x 2 B frags -> 4 wmma per k-step.
// ---------------------------------------------------------------------------
__global__ __launch_bounds__(256)
void gemm_bias_bf16(const bf16_t* __restrict__ A, const bf16_t* __restrict__ W,
                    const float* __restrict__ bias, float* __restrict__ C,
                    int M, int N, int K) {
  __shared__ __align__(16) bf16_t As[64 * 32];    // A tile  [m][k]
  __shared__ __align__(16) bf16_t Bs[128 * 32];   // W tile transposed: [n][k]
  const int tid  = threadIdx.x;
  const int lane = tid & 31;
  const int w    = tid >> 5;
  const int wr   = w >> 2, wc = w & 3;
  const int m0   = blockIdx.x * 64;
  const int n0   = blockIdx.y * 128;

  v8f c00 = {}, c01 = {}, c10 = {}, c11 = {};

  for (int k0 = 0; k0 < K; k0 += 32) {
#if USE_ASYNC_LDS
    {
      const int r = tid >> 2, cq = tid & 3;       // 64 rows x 4 x 16B chunks
      async_copy16((const char*)A + ((size_t)(m0 + r) * K + k0) * 2 + cq * 16,
                   (char*)As + r * 64 + cq * 16);
    }
#else
    {
      const unsigned* Au = (const unsigned*)A;
#pragma unroll
      for (int i = 0; i < 4; ++i) {               // 64x32 bf16 as 1024 dwords
        const int e = tid + i * 256;
        const int r = e >> 4, cc = e & 15;
        ((unsigned*)As)[e] = Au[(size_t)(m0 + r) * (K >> 1) + (k0 >> 1) + cc];
      }
    }
#endif
    // W tile: load dword pairs (two adjacent columns), store transposed
#pragma unroll
    for (int i = 0; i < 8; ++i) {                 // 128x32 bf16 = 2048 dwords
      const int e  = tid + i * 256;
      const int cp = e & 63, k = e >> 6;          // cp: column pair 0..63
      union { unsigned u; unsigned short s[2]; } pk;
      pk.u = ((const unsigned*)W)[((size_t)(k0 + k) * N + n0) / 2 + cp];
      Bs[(2 * cp + 0) * 32 + k] = __builtin_bit_cast(bf16_t, pk.s[0]);
      Bs[(2 * cp + 1) * 32 + k] = __builtin_bit_cast(bf16_t, pk.s[1]);
    }
#if USE_ASYNC_LDS
    async_wait0();
#endif
    __syncthreads();
    const v16bf a0 = frag_from_lds(As, wr * 32 + 0  + (lane & 15), lane);
    const v16bf a1 = frag_from_lds(As, wr * 32 + 16 + (lane & 15), lane);
    const v16bf b0 = frag_from_lds(Bs, wc * 32 + 0  + (lane & 15), lane);
    const v16bf b1 = frag_from_lds(Bs, wc * 32 + 16 + (lane & 15), lane);
    c00 = wmma_bf16(a0, b0, c00);
    c01 = wmma_bf16(a0, b1, c01);
    c10 = wmma_bf16(a1, b0, c10);
    c11 = wmma_bf16(a1, b1, c11);
    __syncthreads();
  }

  const int mof = (lane & 16) ? 8 : 0;
  const int nlo = lane & 15;
#pragma unroll
  for (int mt = 0; mt < 2; ++mt) {
#pragma unroll
    for (int nt = 0; nt < 2; ++nt) {
      const v8f& acc = mt ? (nt ? c11 : c10) : (nt ? c01 : c00);
      const int n = n0 + wc * 32 + nt * 16 + nlo;
      const float bv = bias[n];
      float* cp = C + (size_t)(m0 + wr * 32 + mt * 16 + mof) * N + n;
#pragma unroll
      for (int r = 0; r < 8; ++r) cp[(size_t)r * N] = acc[r] + bv;
    }
  }
}

// ---------------------------------------------------------------------------
// One LSTM timestep, fused: z = h_prev @ Wh (WMMA) + xw[:,t,:]; gates; c/h.
// grid.x = H/16 = 64; each block owns columns [j0, j0+16) for ALL 4 gates and
// all B=64 rows, so the nonlinear cell update is local to the block.
// Wave w: gate g = w>>1, m-tile pair mt = (w&1)*2 .. +1  (16 tiles / 8 waves).
// ---------------------------------------------------------------------------
__global__ __launch_bounds__(256)
void lstm_step(const bf16_t* __restrict__ h_src, bf16_t* __restrict__ h_dst,
               const bf16_t* __restrict__ Wh, const float* __restrict__ xw,
               float* __restrict__ cbuf, float* __restrict__ hf32,
               bf16_t* __restrict__ hs, int t) {
  __shared__ __align__(16) bf16_t As[64 * 32];
  __shared__ __align__(16) bf16_t Bs[64 * 32];
  __shared__ float  zl[64 * 64];
  const int tid  = threadIdx.x;
  const int lane = tid & 31;
  const int w    = tid >> 5;
  const int g    = w >> 1;
  const int mt0  = (w & 1) * 2;
  const int j0   = blockIdx.x * 16;

  v8f acc0 = {}; v8f acc1 = {};

  for (int k0 = 0; k0 < HH; k0 += 32) {
#if USE_ASYNC_LDS
    {
      const int r = tid >> 2, cq = tid & 3;
      async_copy16((const char*)h_src + (size_t)r * (HH * 2) + k0 * 2 + cq * 16,
                   (char*)As + r * 64 + cq * 16);
    }
#else
    {
      const unsigned* Hu = (const unsigned*)h_src;
#pragma unroll
      for (int i = 0; i < 4; ++i) {
        const int e = tid + i * 256;
        const int r = e >> 4, cc = e & 15;
        ((unsigned*)As)[e] = Hu[r * (HH / 2) + (k0 >> 1) + cc];
      }
    }
#endif
    // Wh tile, gate-strided columns, dword-pair loads, transpose store
#pragma unroll
    for (int i = 0; i < 4; ++i) {                 // 64x32 bf16 = 1024 dwords
      const int e  = tid + i * 256;
      const int k  = e >> 5;                      // 0..31
      const int cp = e & 31;                      // column pair 0..31
      const int gg = cp >> 3;                     // gate 0..3
      const int jj = (2 * cp) & 15;
      union { unsigned u; unsigned short s[2]; } pk;
      pk.u = ((const unsigned*)Wh)[((size_t)(k0 + k) * FOURH + gg * HH + j0 + jj) / 2];
      Bs[(2 * cp + 0) * 32 + k] = __builtin_bit_cast(bf16_t, pk.s[0]);
      Bs[(2 * cp + 1) * 32 + k] = __builtin_bit_cast(bf16_t, pk.s[1]);
    }
#if USE_ASYNC_LDS
    async_wait0();
#endif
    __syncthreads();
    const v16bf b  = frag_from_lds(Bs, g * 16 + (lane & 15), lane);
    const v16bf a0 = frag_from_lds(As, (mt0 + 0) * 16 + (lane & 15), lane);
    const v16bf a1 = frag_from_lds(As, (mt0 + 1) * 16 + (lane & 15), lane);
    acc0 = wmma_bf16(a0, b, acc0);
    acc1 = wmma_bf16(a1, b, acc1);
    __syncthreads();
  }

  const int nz  = g * 16 + (lane & 15);
  const int mof = (lane & 16) ? 8 : 0;
#pragma unroll
  for (int r = 0; r < 8; ++r) {
    zl[((mt0 + 0) * 16 + mof + r) * 64 + nz] = acc0[r];
    zl[((mt0 + 1) * 16 + mof + r) * 64 + nz] = acc1[r];
  }
  __syncthreads();

#pragma unroll
  for (int i = 0; i < 4; ++i) {
    const int e  = tid + i * 256;      // 0..1023 = 64 rows x 16 cols
    const int m  = e >> 4;             // batch
    const int jj = e & 15;
    const int col = j0 + jj;
    const float* xr = xw + (size_t)(m * TT + t) * FOURH;
    const float zi = zl[m * 64 +  0 + jj] + xr[col];
    const float zf = zl[m * 64 + 16 + jj] + xr[HH + col];
    const float zg = zl[m * 64 + 32 + jj] + xr[2 * HH + col];
    const float zo = zl[m * 64 + 48 + jj] + xr[3 * HH + col];
    const float si = 1.f / (1.f + __expf(-zi));
    const float sf = 1.f / (1.f + __expf(-zf));
    const float so = 1.f / (1.f + __expf(-zo));
    const float gg = fmaxf(zg, 0.f);
    const float cn = sf * cbuf[m * HH + col] + si * gg;
    const float hn = so * fmaxf(cn, 0.f);
    cbuf[m * HH + col] = cn;
    hf32[m * HH + col] = hn;
    h_dst[m * HH + col] = (bf16_t)hn;
    hs[(size_t)(m * TT + t) * HH + col] = (bf16_t)hn;
  }
}

// ---------------------------------------------------------------------------
// Helpers: f32->bf16 convert, f32 copy, fused concat builder, row softmax.
// ---------------------------------------------------------------------------
__global__ void cvt_f32_bf16(const float* __restrict__ s, bf16_t* __restrict__ d, int n) {
  const int i = blockIdx.x * 256 + threadIdx.x;
  if (i < n) d[i] = (bf16_t)s[i];
}

__global__ void copy_f32(const float* __restrict__ s, float* __restrict__ d, int n) {
  const int i = blockIdx.x * 256 + threadIdx.x;
  if (i < n) d[i] = s[i];
}

__global__ void build_xcat(const float* __restrict__ enc, const float* __restrict__ word,
                           const int* __restrict__ spk, const float* __restrict__ persona,
                           bf16_t* __restrict__ xcat) {
  const size_t i = (size_t)blockIdx.x * 256 + threadIdx.x;
  if (i >= (size_t)MROWS * DIN) return;
  const int k  = (int)(i % DIN);
  const int bt = (int)(i / DIN);
  const int b  = bt / TT;
  float v;
  if (k < HH)            v = enc[(size_t)bt * HH + k];
  else if (k < 2 * HH)   v = word[(size_t)bt * HH + (k - HH)];
  else                   v = persona[(size_t)spk[b] * HH + (k - 2 * HH)];
  xcat[i] = (bf16_t)v;
}

__global__ __launch_bounds__(256)
void softmax_rows(const float* __restrict__ logits, float* __restrict__ out) {
  __shared__ float red[256];
  const int row = blockIdx.x;
  const int tid = threadIdx.x;
  const float* lr = logits + (size_t)row * HH;
  float mx = -3.4e38f;
  for (int i = tid; i < HH; i += 256) mx = fmaxf(mx, lr[i]);
  red[tid] = mx; __syncthreads();
  for (int s = 128; s > 0; s >>= 1) { if (tid < s) red[tid] = fmaxf(red[tid], red[tid + s]); __syncthreads(); }
  mx = red[0]; __syncthreads();
  float vals[4]; float sum = 0.f;
#pragma unroll
  for (int i = 0; i < 4; ++i) { vals[i] = __expf(lr[tid + i * 256] - mx); sum += vals[i]; }
  red[tid] = sum; __syncthreads();
  for (int s = 128; s > 0; s >>= 1) { if (tid < s) red[tid] += red[tid + s]; __syncthreads(); }
  const float inv = 1.f / red[0];
#pragma unroll
  for (int i = 0; i < 4; ++i) out[(size_t)row * HH + tid + i * 256] = vals[i] * inv;
}

// ---------------------------------------------------------------------------
// Host orchestration
// ---------------------------------------------------------------------------
extern "C" void kernel_launch(void* const* d_in, const int* in_sizes, int n_in,
                              void* d_out, int out_size, void* d_ws, size_t ws_size,
                              hipStream_t stream) {
  (void)in_sizes; (void)n_in; (void)out_size; (void)ws_size;
  const float* enc     = (const float*)d_in[0];
  const float* word    = (const float*)d_in[1];
  const int*   spk     = (const int*)  d_in[2];
  const float* h0      = (const float*)d_in[3];
  const float* c0      = (const float*)d_in[4];
  const float* persona = (const float*)d_in[5];
  const float* Wx[4] = {(const float*)d_in[6],  (const float*)d_in[9],
                        (const float*)d_in[12], (const float*)d_in[15]};
  const float* Wh[4] = {(const float*)d_in[7],  (const float*)d_in[10],
                        (const float*)d_in[13], (const float*)d_in[16]};
  const float* bg[4] = {(const float*)d_in[8],  (const float*)d_in[11],
                        (const float*)d_in[14], (const float*)d_in[17]};
  const float* Wd = (const float*)d_in[18];
  const float* bd = (const float*)d_in[19];
  float* out = (float*)d_out;

  // workspace carve-up
  char* wp = (char*)d_ws;
  auto carve = [&](size_t bytes) { char* p = wp; wp += (bytes + 255) & ~(size_t)255; return p; };
  bf16_t* xcat = (bf16_t*)carve((size_t)MROWS * DIN * 2);
  bf16_t* hsA  = (bf16_t*)carve((size_t)MROWS * HH * 2);
  bf16_t* hsB  = (bf16_t*)carve((size_t)MROWS * HH * 2);
  float*  xw   = (float*) carve((size_t)MROWS * FOURH * 4);
  bf16_t* Wxb[4]; bf16_t* Whb[4];
  const int KxL[4] = {DIN, HH, HH, HH};
  for (int l = 0; l < 4; ++l) Wxb[l] = (bf16_t*)carve((size_t)KxL[l] * FOURH * 2);
  for (int l = 0; l < 4; ++l) Whb[l] = (bf16_t*)carve((size_t)HH * FOURH * 2);
  bf16_t* Wdb  = (bf16_t*)carve((size_t)HH * HH * 2);
  bf16_t* hbf0 = (bf16_t*)carve((size_t)BB * HH * 2);
  bf16_t* hbf1 = (bf16_t*)carve((size_t)BB * HH * 2);
  float*  hf32 = (float*) carve((size_t)BB * HH * 4);
  float*  cbuf = (float*) carve((size_t)BB * HH * 4);

  auto nblk = [](size_t n) { return (unsigned)((n + 255) / 256); };

  // 1) convert weights to bf16, build concatenated bf16 input
  for (int l = 0; l < 4; ++l) {
    cvt_f32_bf16<<<nblk((size_t)KxL[l] * FOURH), 256, 0, stream>>>(Wx[l], Wxb[l], KxL[l] * FOURH);
    cvt_f32_bf16<<<nblk((size_t)HH * FOURH), 256, 0, stream>>>(Wh[l], Whb[l], HH * FOURH);
  }
  cvt_f32_bf16<<<nblk((size_t)HH * HH), 256, 0, stream>>>(Wd, Wdb, HH * HH);
  build_xcat<<<nblk((size_t)MROWS * DIN), 256, 0, stream>>>(enc, word, spk, persona, xcat);

  // 2) init recurrent state (state threads through the 4 layers, like the ref)
  copy_f32<<<nblk(BB * HH), 256, 0, stream>>>(h0, hf32, BB * HH);
  copy_f32<<<nblk(BB * HH), 256, 0, stream>>>(c0, cbuf, BB * HH);
  cvt_f32_bf16<<<nblk(BB * HH), 256, 0, stream>>>(h0, hbf0, BB * HH);

  // 3) four LSTM layers: hoisted input-projection GEMM, then fused scan
  const bf16_t* lin[4]  = {xcat, hsA, hsB, hsA};
  bf16_t*       lout[4] = {hsA, hsB, hsA, hsB};
  bf16_t*       hbf[2]  = {hbf0, hbf1};
  for (int l = 0; l < 4; ++l) {
    gemm_bias_bf16<<<dim3(MROWS / 64, FOURH / 128), 256, 0, stream>>>(
        lin[l], Wxb[l], bg[l], xw, MROWS, FOURH, KxL[l]);
    for (int t = 0; t < TT; ++t) {
      lstm_step<<<HH / 16, 256, 0, stream>>>(
          hbf[t & 1], hbf[(t & 1) ^ 1], Whb[l], xw, cbuf, hf32, lout[l], t);
    }
    // T is even -> final h lands back in hbf0, ready for the next layer
  }

  // 4) dense + softmax (logits reuse the xw scratch), then final h/c outputs
  gemm_bias_bf16<<<dim3(MROWS / 64, HH / 128), 256, 0, stream>>>(
      hsB, Wdb, bd, xw, MROWS, HH, HH);
  softmax_rows<<<MROWS, 256, 0, stream>>>(xw, out);
  copy_f32<<<nblk(BB * HH), 256, 0, stream>>>(hf32, out + (size_t)MROWS * HH, BB * HH);
  copy_f32<<<nblk(BB * HH), 256, 0, stream>>>(cbuf, out + (size_t)MROWS * HH + BB * HH, BB * HH);
}
